// Block_11295763989024
// MI455X (gfx1250) — matrix operand order repaired and checked by reading, not compiled, not run on hardware
//
#include <hip/hip_runtime.h>
#include <hip/hip_bf16.h>
#include <math.h>

typedef __attribute__((ext_vector_type(16))) _Float16 v16h;
typedef __attribute__((ext_vector_type(8)))  float    v8f;

__device__ __forceinline__ float gelu_f(float x) {
  return 0.5f * x * (1.0f + erff(x * 0.7071067811865476f));
}
__device__ __forceinline__ float qgelu_f(float x) {
  return x / (1.0f + __expf(-1.702f * x));
}
__device__ __forceinline__ int imin(int a, int b) { return a < b ? a : b; }

// ---------------------------------------------------------------------------
// WMMA GEMM: C[M,N] = postop( [gelu](A[M,K]) @ W[N,K]^T + bias[N] ) [+res]
// Requirements (all call sites satisfy): K % 32 == 0, N % 64 == 0.
// Template: ACT 0=none 1=gelu 2=qgelu (after bias); AGELU: gelu on A load;
// RES 0=none 1=C=res+y 2=C=res+alpha*y.
// 256 thr = 8 waves; tile 128x64; K-chunk 32; f16 WMMA, f32 accum.
// ---------------------------------------------------------------------------
#define BM 128
#define BN 64
#define BK 32

template <int ACT, int AGELU, int RES>
__global__ void __launch_bounds__(256)
k_gemm_wmma(const float* __restrict__ A, const float* __restrict__ W,
            const float* __restrict__ bias, const float* __restrict__ res,
            float* __restrict__ C, int M, int N, int K, float alpha)
{
  __shared__ _Float16 As[BM][BK + 4];
  __shared__ _Float16 Bs[BN][BK + 4];

  const int tid  = threadIdx.x;
  const int lane = tid & 31;
  const int wave = tid >> 5;
  const int tm   = blockIdx.y * BM;
  const int tn   = blockIdx.x * BN;
  const int lr   = lane & 15;
  const int kh   = lane >> 4;
  const int arow = wave * 16 + lr;

  v8f acc[4] = {};

  for (int k0 = 0; k0 < K; k0 += BK) {
    // stage A (128x32): float4 loads -> f16 LDS
#pragma unroll
    for (int i = tid; i < BM * (BK / 4); i += 256) {
      int r  = i >> 3;
      int kq = (i & 7) * 4;
      int gm = imin(tm + r, M - 1);
      const float4 v4 = *(const float4*)(A + (size_t)gm * K + k0 + kq);
      float x0 = v4.x, x1 = v4.y, x2 = v4.z, x3 = v4.w;
      if (AGELU) { x0 = gelu_f(x0); x1 = gelu_f(x1); x2 = gelu_f(x2); x3 = gelu_f(x3); }
      As[r][kq] = (_Float16)x0; As[r][kq + 1] = (_Float16)x1;
      As[r][kq + 2] = (_Float16)x2; As[r][kq + 3] = (_Float16)x3;
      if ((i & 7) == 0 && k0 + BK < K)
        __builtin_prefetch(A + (size_t)gm * K + k0 + BK, 0, 3);  // global_prefetch_b8
    }
    // stage W (64x32)
#pragma unroll
    for (int i = tid; i < BN * (BK / 4); i += 256) {
      int r  = i >> 3;
      int kq = (i & 7) * 4;
      const float4 v4 = *(const float4*)(W + (size_t)(tn + r) * K + k0 + kq);
      Bs[r][kq] = (_Float16)v4.x; Bs[r][kq + 1] = (_Float16)v4.y;
      Bs[r][kq + 2] = (_Float16)v4.z; Bs[r][kq + 3] = (_Float16)v4.w;
      if ((i & 7) == 0 && k0 + BK < K)
        __builtin_prefetch(W + (size_t)(tn + r) * K + k0 + BK, 0, 3);
    }
    __syncthreads();

    v16h afrag;
#pragma unroll
    for (int i = 0; i < 8; i++) {
      int j0 = (i < 4) ? (2 * i + 8 * kh) : (16 + 2 * (i - 4) + 8 * kh);
      afrag[2 * i]     = As[arow][j0];
      afrag[2 * i + 1] = As[arow][j0 + 1];
    }
#pragma unroll
    for (int c = 0; c < 4; c++) {
      v16h bfrag;
      int col = c * 16 + lr;
#pragma unroll
      for (int j = 0; j < 16; j++) bfrag[j] = Bs[col][(j < 8 ? j : j + 8) + 8 * kh];
      acc[c] = __builtin_amdgcn_wmma_f32_16x16x32_f16(
          false, afrag, false, bfrag, (short)0, acc[c], false, false);
    }
    __syncthreads();
  }

#pragma unroll
  for (int c = 0; c < 4; c++) {
    int n = tn + c * 16 + lr;
    int mbase = tm + wave * 16 + kh * 8;
    float bv = (bias != nullptr) ? bias[n] : 0.0f;
#pragma unroll
    for (int r = 0; r < 8; r++) {
      int m = mbase + r;
      if (m < M) {
        float v = acc[c][r] + bv;
        if (ACT == 1)      v = gelu_f(v);
        else if (ACT == 2) v = qgelu_f(v);
        if (RES == 1)      v = res[(size_t)m * N + n] + v;
        else if (RES == 2) v = res[(size_t)m * N + n] + alpha * v;
        C[(size_t)m * N + n] = v;
      }
    }
  }
}

// ---------------------------------------------------------------------------
// Flash attention with WMMA (HD = 32 or 64). Q,K,V,O: [BH, L, HD] contiguous.
// Block = 256 thr = 8 waves; Q-tile 128 rows (16/wave); key tiles of 64.
// Online softmax: per-lane running m/l for 8 rows, 16-lane shfl reductions.
// P re-staged to LDS f16 as WMMA A-fragments; V as B-fragments.
// ---------------------------------------------------------------------------
template <int HD>
__global__ void __launch_bounds__(256)
k_fattn(const float* __restrict__ Q, const float* __restrict__ K,
        const float* __restrict__ V, float* __restrict__ O,
        int Lq, int Lk, float scale)
{
  constexpr int KC = HD / 32;   // K-chunks for Q@K^T
  constexpr int CT = HD / 16;   // 16-col tiles of O
  __shared__ _Float16 Ks[64][HD + 4];
  __shared__ _Float16 Vs[64][HD + 4];
  __shared__ _Float16 Ps[128][64 + 8];

  const int tid  = threadIdx.x;
  const int lane = tid & 31;
  const int wave = tid >> 5;
  const int lr   = lane & 15;
  const int kh   = lane >> 4;
  const int bh   = blockIdx.x;
  const int q0   = blockIdx.y * 128;
  const int mb   = wave * 16 + kh * 8;       // local row base for C/D layout

  const float* Qb = Q + (size_t)bh * Lq * HD;
  const float* Kb = K + (size_t)bh * Lk * HD;
  const float* Vb = V + (size_t)bh * Lk * HD;

  // Q fragments (A layout), pre-scaled, held in registers for all key tiles
  v16h qf[KC];
  {
    int qrow = imin(q0 + wave * 16 + lr, Lq - 1);
    const float* qp = Qb + (size_t)qrow * HD;
#pragma unroll
    for (int kc = 0; kc < KC; kc++)
#pragma unroll
      for (int i = 0; i < 8; i++) {
        int j0 = (i < 4) ? (2 * i + 8 * kh) : (16 + 2 * (i - 4) + 8 * kh);
        qf[kc][2 * i]     = (_Float16)(qp[kc * 32 + j0] * scale);
        qf[kc][2 * i + 1] = (_Float16)(qp[kc * 32 + j0 + 1] * scale);
      }
  }

  float mrun[8], lrun[8];
#pragma unroll
  for (int r = 0; r < 8; r++) { mrun[r] = -1e30f; lrun[r] = 0.0f; }
  v8f accO[CT] = {};

  for (int kt = 0; kt < Lk; kt += 64) {
    // stage K,V tiles (64 x HD)
#pragma unroll
    for (int i = tid; i < 64 * (HD / 4); i += 256) {
      int r  = i / (HD / 4);
      int cq = (i % (HD / 4)) * 4;
      int gr = imin(kt + r, Lk - 1);
      const float4 k4 = *(const float4*)(Kb + (size_t)gr * HD + cq);
      Ks[r][cq] = (_Float16)k4.x; Ks[r][cq + 1] = (_Float16)k4.y;
      Ks[r][cq + 2] = (_Float16)k4.z; Ks[r][cq + 3] = (_Float16)k4.w;
      const float4 v4 = *(const float4*)(Vb + (size_t)gr * HD + cq);
      Vs[r][cq] = (_Float16)v4.x; Vs[r][cq + 1] = (_Float16)v4.y;
      Vs[r][cq + 2] = (_Float16)v4.z; Vs[r][cq + 3] = (_Float16)v4.w;
    }
    __syncthreads();

    // scores: S = Q @ K^T  (128 x 64 per block; 4 col tiles per wave)
    v8f accS[4] = {};
#pragma unroll
    for (int kc = 0; kc < KC; kc++) {
#pragma unroll
      for (int c = 0; c < 4; c++) {
        v16h bfrag;
        int col = c * 16 + lr;                 // key within tile
#pragma unroll
        for (int j = 0; j < 16; j++) bfrag[j] = Ks[col][kc * 32 + 8 * kh + (j < 8 ? j : j + 8)];
        accS[c] = __builtin_amdgcn_wmma_f32_16x16x32_f16(
            false, qf[kc], false, bfrag, (short)0, accS[c], false, false);
      }
    }

    // online softmax for the 8 rows this lane owns (shared by 16-lane group)
    float tmax[8];
#pragma unroll
    for (int r = 0; r < 8; r++) {
      float t = -1e30f;
#pragma unroll
      for (int c = 0; c < 4; c++) {
        int col = kt + c * 16 + lr;
        float s = (col < Lk) ? accS[c][r] : -1e30f;
        t = fmaxf(t, s);
      }
      tmax[r] = t;
    }
#pragma unroll
    for (int off = 1; off < 16; off <<= 1)
#pragma unroll
      for (int r = 0; r < 8; r++)
        tmax[r] = fmaxf(tmax[r], __shfl_xor(tmax[r], off, 32));

    float psum[8];
#pragma unroll
    for (int r = 0; r < 8; r++) {
      float newm = fmaxf(mrun[r], tmax[r]);
      float ps = 0.0f;
#pragma unroll
      for (int c = 0; c < 4; c++) {
        int col = kt + c * 16 + lr;
        float e = (col < Lk) ? __expf(accS[c][r] - newm) : 0.0f;
        Ps[mb + r][c * 16 + lr] = (_Float16)e;
        ps += e;
      }
      float rescale = __expf(mrun[r] - newm);
      mrun[r] = newm;
      psum[r] = ps;
      // rescale running output accumulator
#pragma unroll
      for (int c = 0; c < CT; c++) accO[c][r] *= rescale;
      lrun[r] *= rescale;
    }
#pragma unroll
    for (int off = 1; off < 16; off <<= 1)
#pragma unroll
      for (int r = 0; r < 8; r++) psum[r] += __shfl_xor(psum[r], off, 32);
#pragma unroll
    for (int r = 0; r < 8; r++) lrun[r] += psum[r];

    // O += P @ V   (P rows are wave-local in LDS; same-wave DS ops are ordered)
#pragma unroll
    for (int kc = 0; kc < 2; kc++) {          // 64 keys = 2 chunks of 32
      v16h pf;
      int prow = wave * 16 + lr;
#pragma unroll
      for (int i = 0; i < 8; i++) {
        int j0 = (i < 4) ? (2 * i + 8 * kh) : (16 + 2 * (i - 4) + 8 * kh);
        pf[2 * i]     = Ps[prow][kc * 32 + j0];
        pf[2 * i + 1] = Ps[prow][kc * 32 + j0 + 1];
      }
#pragma unroll
      for (int c = 0; c < CT; c++) {
        v16h bfrag;
        int col = c * 16 + lr;                 // output dim
#pragma unroll
        for (int j = 0; j < 16; j++) bfrag[j] = Vs[kc * 32 + 8 * kh + (j < 8 ? j : j + 8)][col];
        accO[c] = __builtin_amdgcn_wmma_f32_16x16x32_f16(
            false, pf, false, bfrag, (short)0, accO[c], false, false);
      }
    }
    __syncthreads();
  }

  // epilogue: normalize by l and store
#pragma unroll
  for (int c = 0; c < CT; c++) {
    int n = c * 16 + lr;
#pragma unroll
    for (int r = 0; r < 8; r++) {
      int m = q0 + mb + r;
      if (m < Lq)
        O[((size_t)bh * Lq + m) * HD + n] = accO[c][r] / lrun[r];
    }
  }
}

// ---------------------------------------------------------------------------
// LayerNorm over last dim (one block per row)
// ---------------------------------------------------------------------------
__global__ void __launch_bounds__(256)
k_layernorm(const float* __restrict__ x, const float* __restrict__ g,
            const float* __restrict__ b, float* __restrict__ y, int D)
{
  __shared__ float red[256];
  int row = blockIdx.x;
  const float* xr = x + (size_t)row * D;
  float s = 0.0f;
  for (int i = threadIdx.x; i < D; i += 256) s += xr[i];
  red[threadIdx.x] = s; __syncthreads();
  for (int o = 128; o > 0; o >>= 1) {
    if (threadIdx.x < o) red[threadIdx.x] += red[threadIdx.x + o];
    __syncthreads();
  }
  float mean = red[0] / (float)D; __syncthreads();
  float v = 0.0f;
  for (int i = threadIdx.x; i < D; i += 256) { float d = xr[i] - mean; v += d * d; }
  red[threadIdx.x] = v; __syncthreads();
  for (int o = 128; o > 0; o >>= 1) {
    if (threadIdx.x < o) red[threadIdx.x] += red[threadIdx.x + o];
    __syncthreads();
  }
  float rstd = rsqrtf(red[0] / (float)D + 1e-5f);
  float* yr = y + (size_t)row * D;
  for (int i = threadIdx.x; i < D; i += 256)
    yr[i] = (xr[i] - mean) * rstd * g[i] + b[i];
}

// ---------------------------------------------------------------------------
// VALU attention for tiny L (L=8 cases). Q,K,V,O: [BH, L, hd].
// ---------------------------------------------------------------------------
__global__ void __launch_bounds__(128)
k_attn(const float* __restrict__ Q, const float* __restrict__ K,
       const float* __restrict__ V, float* __restrict__ O,
       int Lq, int Lk, int hd, float scale)
{
  extern __shared__ float sm[];
  __shared__ float red[128];
  int bh = blockIdx.x;
  int q  = blockIdx.y;
  float* sc = sm;
  float* qv = sm + Lk;
  const float* qp = Q + ((size_t)bh * Lq + q) * hd;
  for (int i = threadIdx.x; i < hd; i += 128) qv[i] = qp[i] * scale;
  __syncthreads();
  const float* kb = K + (size_t)bh * Lk * hd;
  for (int kk = threadIdx.x; kk < Lk; kk += 128) {
    const float* kp = kb + (size_t)kk * hd;
    float d = 0.0f;
    for (int i = 0; i < hd; i++) d += qv[i] * kp[i];
    sc[kk] = d;
  }
  __syncthreads();
  float mx = -1e30f;
  for (int kk = threadIdx.x; kk < Lk; kk += 128) mx = fmaxf(mx, sc[kk]);
  red[threadIdx.x] = mx; __syncthreads();
  for (int o = 64; o > 0; o >>= 1) {
    if (threadIdx.x < o) red[threadIdx.x] = fmaxf(red[threadIdx.x], red[threadIdx.x + o]);
    __syncthreads();
  }
  mx = red[0]; __syncthreads();
  float sum = 0.0f;
  for (int kk = threadIdx.x; kk < Lk; kk += 128) {
    float e = __expf(sc[kk] - mx); sc[kk] = e; sum += e;
  }
  red[threadIdx.x] = sum; __syncthreads();
  for (int o = 64; o > 0; o >>= 1) {
    if (threadIdx.x < o) red[threadIdx.x] += red[threadIdx.x + o];
    __syncthreads();
  }
  float inv = 1.0f / red[0];
  __syncthreads();
  const float* vb = V + (size_t)bh * Lk * hd;
  float* op = O + ((size_t)bh * Lq + q) * hd;
  for (int i = threadIdx.x; i < hd; i += 128) {
    float a = 0.0f;
    for (int kk = 0; kk < Lk; kk++) a += sc[kk] * vb[(size_t)kk * hd + i];
    op[i] = a * inv;
  }
}

// ---------------------------------------------------------------------------
// Head split / merge / permute / elementwise
// ---------------------------------------------------------------------------
__global__ void k_split_heads(const float* __restrict__ in, float* __restrict__ o0,
                              float* __restrict__ o1, float* __restrict__ o2,
                              int S, int Bn, int nmat, int H, int hd, int seq_major)
{
  size_t total = (size_t)S * Bn * nmat * H * hd;
  size_t idx = (size_t)blockIdx.x * blockDim.x + threadIdx.x;
  if (idx >= total) return;
  int d = idx % hd;   size_t t = idx / hd;
  int s = t % S;      t /= S;
  int h = t % H;      t /= H;
  int b = t % Bn;     int mat = (int)(t / Bn);
  size_t token = seq_major ? ((size_t)s * Bn + b) : ((size_t)b * S + s);
  float v = in[token * (size_t)(nmat * H * hd) + (size_t)mat * H * hd + (size_t)h * hd + d];
  float* dst = (mat == 0) ? o0 : ((mat == 1) ? o1 : o2);
  dst[(((size_t)b * H + h) * S + s) * hd + d] = v;
}

__global__ void k_merge_heads(const float* __restrict__ O, float* __restrict__ out,
                              int S, int Bn, int H, int hd, int seq_major)
{
  size_t total = (size_t)S * Bn * H * hd;
  size_t idx = (size_t)blockIdx.x * blockDim.x + threadIdx.x;
  if (idx >= total) return;
  int d = idx % hd;   size_t t = idx / hd;
  int s = t % S;      t /= S;
  int h = t % H;      int b = (int)(t / H);
  size_t token = seq_major ? ((size_t)s * Bn + b) : ((size_t)b * S + s);
  out[token * (size_t)(H * hd) + (size_t)h * hd + d] =
      O[(((size_t)b * H + h) * S + s) * hd + d];
}

__global__ void k_perm4(const float* __restrict__ src, float* __restrict__ dst,
                        int d0, int d1, int d2, int d3,
                        int p0, int p1, int p2, int p3)
{
  size_t total = (size_t)d0 * d1 * d2 * d3;
  size_t idx = (size_t)blockIdx.x * blockDim.x + threadIdx.x;
  if (idx >= total) return;
  int sd[4] = { d0, d1, d2, d3 };
  int e1 = sd[p1], e2 = sd[p2], e3 = sd[p3];
  int c3 = idx % e3;  size_t t = idx / e3;
  int c2 = t % e2;    t /= e2;
  int c1 = t % e1;    int c0 = (int)(t / e1);
  int sc[4];
  sc[p0] = c0; sc[p1] = c1; sc[p2] = c2; sc[p3] = c3;
  size_t si = (((size_t)sc[0] * d1 + sc[1]) * d2 + sc[2]) * d3 + sc[3];
  dst[idx] = src[si];
}

__global__ void k_add(float* __restrict__ y, const float* __restrict__ a,
                      const float* __restrict__ b, size_t n)
{
  size_t i = (size_t)blockIdx.x * blockDim.x + threadIdx.x;
  if (i < n) y[i] = a[i] + b[i];
}

__global__ void k_add_pos(float* __restrict__ x, const float* __restrict__ pos,
                          int outer, int mid, int inner)
{
  size_t n = (size_t)outer * mid * inner;
  size_t i = (size_t)blockIdx.x * blockDim.x + threadIdx.x;
  if (i >= n) return;
  int c = i % inner;
  int m = (int)((i / inner) % mid);
  x[i] += pos[(size_t)m * inner + c];
}

__global__ void k_copy(float* __restrict__ dst, const float* __restrict__ src, size_t n)
{
  size_t i = (size_t)blockIdx.x * blockDim.x + threadIdx.x;
  if (i < n) dst[i] = src[i];
}

// ---------------------------------------------------------------------------
// GEMM dispatch over template variants actually used
// ---------------------------------------------------------------------------
static void gemm_launch(const float* A, const float* W, const float* bias,
                        const float* res, float* C, int M, int N, int K,
                        int act, int ag, int rm, float alpha, hipStream_t st)
{
  dim3 g(N / BN, (M + BM - 1) / BM);
  if (ag)            k_gemm_wmma<0,1,1><<<g, 256, 0, st>>>(A, W, bias, res, C, M, N, K, alpha);
  else if (act == 1) k_gemm_wmma<1,0,0><<<g, 256, 0, st>>>(A, W, bias, res, C, M, N, K, alpha);
  else if (act == 2) k_gemm_wmma<2,0,0><<<g, 256, 0, st>>>(A, W, bias, res, C, M, N, K, alpha);
  else if (rm == 1)  k_gemm_wmma<0,0,1><<<g, 256, 0, st>>>(A, W, bias, res, C, M, N, K, alpha);
  else if (rm == 2)  k_gemm_wmma<0,0,2><<<g, 256, 0, st>>>(A, W, bias, res, C, M, N, K, alpha);
  else               k_gemm_wmma<0,0,0><<<g, 256, 0, st>>>(A, W, bias, res, C, M, N, K, alpha);
}

// ---------------------------------------------------------------------------
// Orchestration
// ---------------------------------------------------------------------------
extern "C" void kernel_launch(void* const* d_in, const int* in_sizes, int n_in,
                              void* d_out, int out_size, void* d_ws, size_t ws_size,
                              hipStream_t stream)
{
  (void)in_sizes; (void)out_size; (void)ws_size;
  const int D = 768, D2 = 384, H = 12, AD = 192;
  const int NS  = 197 * 32;            // 6304 s-tokens
  const int NTK = 4 * 1568;            // 6272 t-tokens
  const float sc64 = 0.125f;
  const float sc32 = 0.17677669529663687f;

  const float* IN[80];
  for (int i = 0; i < n_in; i++) IN[i] = (const float*)d_in[i];
  const float* s_x = IN[0];
  const float* t_x = IN[1];

  float* wsp = (float*)d_ws;
  size_t off = 0;
  auto alloc = [&](size_t n) { float* p = wsp + off; off += n; return p; };
  float* big0 = alloc((size_t)NS * 3072);
  float* big1 = alloc((size_t)NS * 3072);
  float* Qb = alloc((size_t)NS * D);
  float* Kb = alloc((size_t)NS * D);
  float* Vb = alloc((size_t)NS * D);
  float* Ob = alloc((size_t)NS * D);
  float* sxa = alloc((size_t)NS * D);
  float* sxb = alloc((size_t)NS * D);
  float* txa = alloc((size_t)NTK * D);
  float* txb = alloc((size_t)NTK * D);
  float* nsb = alloc((size_t)NS * D2);
  float* ntb = alloc((size_t)NTK * D2);
  float* csb = alloc((size_t)NS * D2);
  float* ctb = alloc((size_t)NTK * D2);
  float* pa   = alloc((size_t)NTK * D2);
  float* pbuf = alloc((size_t)NTK * D2);

  float* out_s = (float*)d_out;
  float* out_t = out_s + (size_t)NS * D;

  auto gemm = [&](const float* A, const float* W, const float* bias,
                  const float* res, float* C, int M, int N, int K,
                  int act, int ag, int rm, float alpha) {
    gemm_launch(A, W, bias, res, C, M, N, K, act, ag, rm, alpha, stream);
  };
  auto ln = [&](const float* x, const float* g, const float* b, float* y,
                int rows, int Dd) {
    k_layernorm<<<rows, 256, 0, stream>>>(x, g, b, y, Dd);
  };
  auto attn_small = [&](const float* Q, const float* K, const float* V, float* O,
                        int BH, int Lq, int Lk, int hd, float scale) {
    size_t shm = (size_t)(Lk + hd) * sizeof(float);
    k_attn<<<dim3(BH, Lq), 128, shm, stream>>>(Q, K, V, O, Lq, Lk, hd, scale);
  };
  auto split = [&](const float* in, float* o0, float* o1, float* o2,
                   int S, int Bn, int nmat, int hd, int seqm) {
    size_t tot = (size_t)S * Bn * nmat * H * hd;
    k_split_heads<<<(tot + 255) / 256, 256, 0, stream>>>(in, o0, o1, o2, S, Bn,
                                                         nmat, H, hd, seqm);
  };
  auto merge = [&](const float* O, float* out, int S, int Bn, int hd, int seqm) {
    size_t tot = (size_t)S * Bn * H * hd;
    k_merge_heads<<<(tot + 255) / 256, 256, 0, stream>>>(O, out, S, Bn, H, hd, seqm);
  };
  auto perm4 = [&](const float* s, float* d, int d0, int d1, int d2, int d3,
                   int p0, int p1, int p2, int p3) {
    size_t tot = (size_t)d0 * d1 * d2 * d3;
    k_perm4<<<(tot + 255) / 256, 256, 0, stream>>>(s, d, d0, d1, d2, d3, p0, p1, p2, p3);
  };
  auto vadd = [&](float* y, const float* a, const float* b, size_t n) {
    k_add<<<(n + 255) / 256, 256, 0, stream>>>(y, a, b, n);
  };

  // ---- 1) spatial MHA + adapter on s_x --------------------------------------
  ln(s_x, IN[2], IN[3], big0, NS, D);
  gemm(big0, IN[4], IN[5], nullptr, big1, NS, 3 * D, D, 0, 0, 0, 0.f);
  split(big1, Qb, Kb, Vb, 197, 32, 3, 64, 1);
  k_fattn<64><<<dim3(32 * H, 2), 256, 0, stream>>>(Qb, Kb, Vb, Ob, 197, 197, sc64);
  merge(Ob, big0, 197, 32, 64, 1);
  gemm(big0, IN[6], IN[7], nullptr, big1, NS, D, D, 0, 0, 0, 0.f);     // mha out
  gemm(big1, IN[14], IN[15], nullptr, big0, NS, AD, D, 1, 0, 0, 0.f);  // gelu a1
  gemm(big0, IN[16], IN[17], big1, Ob, NS, D, AD, 0, 0, 1, 0.f);       // adapter
  vadd(sxa, s_x, Ob, (size_t)NS * D);                                  // s_x1

  // ---- 2) temporal MHA (L=8 -> VALU attention) ------------------------------
  perm4(sxa, big0, 197, 4, 8, D, 2, 1, 0, 3);                 // xt (8,788,D)
  ln(big0, IN[8], IN[9], big1, NS, D);
  gemm(big1, IN[10], IN[11], nullptr, big0, NS, 3 * D, D, 0, 0, 0, 0.f);
  split(big0, Qb, Kb, Vb, 8, 788, 3, 64, 1);
  attn_small(Qb, Kb, Vb, Ob, 788 * H, 8, 8, 64, sc64);
  merge(Ob, big0, 8, 788, 64, 1);
  gemm(big0, IN[12], IN[13], nullptr, big1, NS, D, D, 0, 0, 0, 0.f);
  perm4(big1, big0, 8, 4, 197, D, 2, 1, 0, 3);
  vadd(sxb, sxa, big0, (size_t)NS * D);                       // s_x2

  // ---- 3) t_x self-attention + adapter (L=1568 -> flash WMMA) ---------------
  ln(t_x, IN[18], IN[19], big0, NTK, D);
  gemm(big0, IN[20], nullptr, nullptr, big1, NTK, 3 * D, D, 0, 0, 0, 0.f);
  split(big1, Qb, Kb, Vb, 1568, 4, 3, 64, 0);
  k_fattn<64><<<dim3(4 * H, 13), 256, 0, stream>>>(Qb, Kb, Vb, Ob, 1568, 1568, sc64);
  merge(Ob, big0, 1568, 4, 64, 0);
  gemm(big0, IN[21], IN[22], nullptr, big1, NTK, D, D, 0, 0, 0, 0.f);  // proj
  gemm(big1, IN[23], IN[24], nullptr, big0, NTK, AD, D, 1, 0, 0, 0.f);
  gemm(big0, IN[25], IN[26], big1, Ob, NTK, D, AD, 0, 0, 1, 0.f);
  vadd(txa, t_x, Ob, (size_t)NTK * D);                        // t_x1

  // ---- 4) cross-stream ------------------------------------------------------
  gemm(sxb, IN[27], IN[28], nullptr, big0, NS, D2, D, 0, 0, 0, 0.f);
  ln(big0, IN[31], IN[32], nsb, NS, D2);                      // ns
  gemm(txa, IN[29], IN[30], nullptr, big0, NTK, D2, D, 0, 0, 0, 0.f);
  ln(big0, IN[33], IN[34], ntb, NTK, D2);                     // nt

  // t2s (L=8 -> VALU attention)
  perm4(nsb + (size_t)32 * D2, pa, 196, 4, 8, D2, 1, 0, 2, 3);   // sp (784,8,D2)
  k_add_pos<<<((size_t)784 * 8 * D2 + 255) / 256, 256, 0, stream>>>(pa, IN[35], 784, 8, D2);
  perm4(ntb, pbuf, 4, 8, 196, D2, 0, 2, 1, 3);                   // tt (784,8,D2)
  k_add_pos<<<((size_t)784 * 8 * D2 + 255) / 256, 256, 0, stream>>>(pbuf, IN[36], 784, 8, D2);
  gemm(pa, IN[37], IN[38], nullptr, big0, NTK, D2, D2, 0, 0, 0, 0.f);
  split(big0, Qb, nullptr, nullptr, 8, 784, 1, 32, 0);
  gemm(pbuf, IN[39], IN[40], nullptr, big0, NTK, 2 * D2, D2, 0, 0, 0, 0.f);
  split(big0, Kb, Vb, nullptr, 8, 784, 2, 32, 0);
  attn_small(Qb, Kb, Vb, Ob, 784 * H, 8, 8, 32, sc32);
  merge(Ob, big0, 8, 784, 32, 0);
  gemm(big0, IN[41], IN[42], nullptr, big1, NTK, D2, D2, 0, 0, 0, 0.f);
  k_copy<<<((size_t)32 * D2 + 255) / 256, 256, 0, stream>>>(csb, nsb, (size_t)32 * D2);
  perm4(big1, csb + (size_t)32 * D2, 4, 196, 8, D2, 1, 0, 2, 3);  // c_s

  // s2t (L=196 -> flash WMMA, HD=32)
  perm4(nsb + (size_t)32 * D2, pa, 196, 32, D2, 1, 1, 0, 2, 3);   // sp2 (32,196,D2)
  k_add_pos<<<((size_t)32 * 196 * D2 + 255) / 256, 256, 0, stream>>>(pa, IN[43], 32, 196, D2);
  k_copy<<<((size_t)NTK * D2 + 255) / 256, 256, 0, stream>>>(pbuf, ntb, (size_t)NTK * D2);
  k_add_pos<<<((size_t)32 * 196 * D2 + 255) / 256, 256, 0, stream>>>(pbuf, IN[44], 32, 196, D2);
  gemm(pbuf, IN[45], IN[46], nullptr, big0, NTK, D2, D2, 0, 0, 0, 0.f);
  split(big0, Qb, nullptr, nullptr, 196, 32, 1, 32, 0);
  gemm(pa, IN[47], IN[48], nullptr, big0, NTK, 2 * D2, D2, 0, 0, 0, 0.f);
  split(big0, Kb, Vb, nullptr, 196, 32, 2, 32, 0);
  k_fattn<32><<<dim3(32 * H, 2), 256, 0, stream>>>(Qb, Kb, Vb, Ob, 196, 196, sc32);
  merge(Ob, big0, 196, 32, 32, 0);
  gemm(big0, IN[49], IN[50], nullptr, ctb, NTK, D2, D2, 0, 0, 0, 0.f); // c_t

  // ---- 5) fusion updates (gelu fused into GEMM A-load) ----------------------
  gemm(csb, IN[51], IN[52], sxb, sxa, NS, D, D2, 0, 1, 1, 0.f);   // s_x3
  gemm(ctb, IN[53], IN[54], txa, txb, NTK, D, D2, 0, 1, 1, 0.f);  // t_x2

  // ---- 6) s-stream MLP + 0.5*adapter ---------------------------------------
  ln(sxa, IN[59], IN[60], big0, NS, D);                           // xn
  gemm(big0, IN[55], IN[56], nullptr, big1, NS, 4 * D, D, 2, 0, 0, 0.f); // qgelu
  gemm(big1, IN[57], IN[58], sxa, sxb, NS, D, 4 * D, 0, 0, 1, 0.f);
  gemm(big0, IN[61], IN[62], nullptr, big1, NS, AD, D, 1, 0, 0, 0.f);
  gemm(big1, IN[63], IN[64], sxb, out_s, NS, D, AD, 0, 0, 2, 0.5f);

  // ---- 7) t-stream MLP + 0.5*adapter ---------------------------------------
  ln(txb, IN[65], IN[66], big0, NTK, D);
  gemm(big0, IN[67], IN[68], nullptr, big1, NTK, 4 * D, D, 1, 0, 0, 0.f);
  gemm(big1, IN[69], IN[70], txb, txa, NTK, D, 4 * D, 0, 0, 1, 0.f);
  gemm(big0, IN[71], IN[72], nullptr, big1, NTK, AD, D, 1, 0, 0, 0.f);
  gemm(big1, IN[73], IN[74], txa, out_t, NTK, D, AD, 0, 0, 2, 0.5f);
}
